// HetGCN_76682346102819
// MI455X (gfx1250) — matrix operand order repaired and checked by reading, not compiled, not run on hardware
//
#include <hip/hip_runtime.h>
#include <hip/hip_bf16.h>
#include <math.h>

// ---------------------------------------------------------------------------
// CDNA5 (gfx1250) HetGCN forward. Dense GEMMs use v_wmma_f32_16x16x32_bf16
// (bf16 in / f32 accumulate). LDS tiles are stored pre-swizzled so every
// WMMA fragment is exactly two ds_load_b128 per lane; all B fragments of a
// k-step are batch-issued ahead of the WMMA chain (sched_barrier) so LDS
// latency overlaps the matrix pipe. SpMM uses per-edge waves with global
// FP32 atomics (resolved in the 192MB L2). Semantic attention is fused into
// the GEMM epilogue (tanh + q-dot + scalar atomic reduce).
// ---------------------------------------------------------------------------

typedef __attribute__((ext_vector_type(16))) __bf16 v16bf;
typedef __attribute__((ext_vector_type(8)))  __bf16 v8bf;
typedef __attribute__((ext_vector_type(8)))  float  v8f;
typedef __attribute__((ext_vector_type(4)))  unsigned short us4;

#define AS 72    // LDS A-tile row stride in ushorts (multiple of 8 -> 16B align)

__device__ __forceinline__ unsigned short f2bf(float f) {
  union { float f; unsigned int u; } cv; cv.f = f;
  unsigned int u = cv.u;
  u += 0x7fffu + ((u >> 16) & 1u);          // round-to-nearest-even
  return (unsigned short)(u >> 16);
}

__device__ __forceinline__ v16bf bfcat(v8bf lo, v8bf hi) {
  return __builtin_shufflevector(lo, hi, 0, 1, 2, 3, 4, 5, 6, 7,
                                 8, 9, 10, 11, 12, 13, 14, 15);
}

__device__ __forceinline__ void sched_fence() {
#if __has_builtin(__builtin_amdgcn_sched_barrier)
  __builtin_amdgcn_sched_barrier(0);   // keep load batch ahead of WMMA chain
#endif
}

// MODE: 0 = C = A@W + b            (store)
//       1 = C = relu(A@W + b)      (store)
//       2 = ssum += sum_{i,n} tanh((A@W)[i,n] + b[n]) * q[n]   (no store)
// Output width M = NT*16 (compile-time so accumulators stay in VGPRs).
template <int NT, int MODE>
__global__ __launch_bounds__(256)
void gemm_wmma(const float* __restrict__ A, const float* __restrict__ W,
               const float* __restrict__ bias, float* __restrict__ C,
               const float* __restrict__ q, float* __restrict__ ssum,
               int Nrows, int K) {
  constexpr int M = NT * 16;
  // A tile: 128 rows x 64 k, row-major bf16 (each lane's frag = 2x 16B runs)
  __shared__ __attribute__((aligned(16))) unsigned short sA[128 * AS];
  // W tile in fragment-major order: [kstep(2)][tile(NT)][lane(32)][elem(16)]
  __shared__ __attribute__((aligned(16))) unsigned short sWf[2 * NT * 32 * 16];

  const int tid  = threadIdx.x;
  const int wv   = tid >> 5;     // wave id 0..7 (wave32)
  const int lane = tid & 31;
  const int half = lane >> 4;    // 0: lanes 0-15, 1: lanes 16-31
  const int lp   = lane & 15;
  const int rowBase = blockIdx.x * 128;

  v8f acc[NT];
  const v8f vzero = {0.f, 0.f, 0.f, 0.f, 0.f, 0.f, 0.f, 0.f};
#pragma unroll
  for (int t = 0; t < NT; ++t) acc[t] = vzero;

  for (int kk = 0; kk < K; kk += 64) {
    // Stage A tile (128x64) as bf16, float4 loads -> packed ushort4 stores
    for (int idx = tid * 4; idx < 128 * 64; idx += 256 * 4) {
      const int r = idx >> 6, c = idx & 63;  // c is a multiple of 4
      const int gr = rowBase + r;
      float4 v = make_float4(0.f, 0.f, 0.f, 0.f);
      if (gr < Nrows) v = *(const float4*)&A[(size_t)gr * K + kk + c];
      us4 p; p.x = f2bf(v.x); p.y = f2bf(v.y); p.z = f2bf(v.z); p.w = f2bf(v.w);
      *(us4*)&sA[r * AS + c] = p;
    }
    // Stage W tile (64xM) pre-swizzled into fragment-major order.
    // B frag layout: elem j -> K = half*8 + j (j<8), 16 + half*8 + (j-8) (j>=8), N = lane%16.
    for (int idx = tid; idx < 64 * M; idx += 256) {
      const int r = idx / M;            // k within chunk, 0..63
      const int c = idx - r * M;        // output column
      const int ks2 = r >> 5;           // which 32-wide k-step
      const int kr  = r & 31;
      const int jj  = (kr & 7) + ((kr >> 4) << 3);
      const int hh  = (kr >> 3) & 1;
      const int ln  = hh * 16 + (c & 15);
      const int tt  = c >> 4;
      sWf[(((ks2 * NT) + tt) * 32 + ln) * 16 + jj] =
          f2bf(W[(size_t)(kk + r) * M + c]);
    }
    __syncthreads();

#pragma unroll
    for (int ks2 = 0; ks2 < 2; ++ks2) {
      // A fragment: lanes 0-15 row=lp, K elems [kb..kb+7] and [kb+16..kb+23]
      // (kb = half*8); both runs are 16B-aligned contiguous -> 2x ds_load_b128
      const unsigned short* ap = &sA[(wv * 16 + lp) * AS + ks2 * 32 + half * 8];
      const v16bf af = bfcat(*(const v8bf*)ap, *(const v8bf*)(ap + 16));
      // Batch-issue all B fragments of this k-step, then fence the scheduler
      // so the ds_load_b128 batch stays ahead of the WMMA chain (staggered
      // s_wait_dscnt instead of a full stall per WMMA).
      v16bf bfr[NT];
#pragma unroll
      for (int t = 0; t < NT; ++t) {
        const unsigned short* wp = &sWf[(((ks2 * NT) + t) * 32 + lane) * 16];
        bfr[t] = bfcat(*(const v8bf*)wp, *(const v8bf*)(wp + 8));
      }
      sched_fence();
#pragma unroll
      for (int t = 0; t < NT; ++t)
        acc[t] = __builtin_amdgcn_wmma_f32_16x16x32_bf16(
            false, af, false, bfr[t], (short)0, acc[t], false, false);
      sched_fence();
    }
    __syncthreads();
  }

  // Epilogue. C/D layout: vgpr v, lane l -> row = v + (l<16?0:8), col = l%16
  const int rloc = wv * 16 + half * 8;
  if (MODE == 2) {
    float partial = 0.f;
#pragma unroll
    for (int t = 0; t < NT; ++t) {
      const int n  = t * 16 + lp;
      const float bv = bias[n];
      const float qv = q[n];
#pragma unroll
      for (int v = 0; v < 8; ++v) {
        const int gr = rowBase + rloc + v;
        if (gr < Nrows) partial += tanhf(acc[t][v] + bv) * qv;
      }
    }
    __hip_atomic_fetch_add(ssum, partial, __ATOMIC_RELAXED,
                           __HIP_MEMORY_SCOPE_AGENT);
  } else {
#pragma unroll
    for (int t = 0; t < NT; ++t) {
      const int n  = t * 16 + lp;
      const float bv = bias ? bias[n] : 0.f;
#pragma unroll
      for (int v = 0; v < 8; ++v) {
        const int gr = rowBase + rloc + v;
        if (gr < Nrows) {
          float o = acc[t][v] + bv;
          if (MODE == 1) o = fmaxf(o, 0.f);
          C[(size_t)gr * M + n] = o;
        }
      }
    }
  }
}

// h[i,:] = a2[i] * y[i,:]   (initializes SpMM accumulator)
__global__ void scale_rows(const float* __restrict__ a2, const float* __restrict__ y,
                           float* __restrict__ h, int total) {
  const int t = blockIdx.x * blockDim.x + threadIdx.x;
  if (t >= total) return;
  h[t] = a2[t >> 7] * y[t];
}

// h[dst[e],:] += val[e] * x[src[e],:]  — one edge per wave, float4 per lane
__global__ void spmm_edges(const int* __restrict__ src, const int* __restrict__ dst,
                           const float* __restrict__ val, const float* __restrict__ x,
                           float* __restrict__ h, int E) {
  const int t = blockIdx.x * blockDim.x + threadIdx.x;
  const int e = t >> 5;
  if (e >= E) return;
  const int lane = t & 31;
  const int s = src[e], d = dst[e];
  const float v = val[e];
  const float4 xv = ((const float4*)(x + (size_t)s * 128))[lane];
  float* hd = h + (size_t)d * 128 + lane * 4;
  __hip_atomic_fetch_add(hd + 0, v * xv.x, __ATOMIC_RELAXED, __HIP_MEMORY_SCOPE_AGENT);
  __hip_atomic_fetch_add(hd + 1, v * xv.y, __ATOMIC_RELAXED, __HIP_MEMORY_SCOPE_AGENT);
  __hip_atomic_fetch_add(hd + 2, v * xv.z, __ATOMIC_RELAXED, __HIP_MEMORY_SCOPE_AGENT);
  __hip_atomic_fetch_add(hd + 3, v * xv.w, __ATOMIC_RELAXED, __HIP_MEMORY_SCOPE_AGENT);
}

// beta = softmax(ssum/N over 2 relations); out = relu(b0*h0 + b1*h1)
__global__ void softmax2_combine(const float* __restrict__ ss, const float* __restrict__ h0,
                                 const float* __restrict__ h1, float* __restrict__ out,
                                 int total, float invN) {
  const int t = blockIdx.x * blockDim.x + threadIdx.x;
  if (t >= total) return;
  const float s0 = ss[0] * invN, s1 = ss[1] * invN;
  const float m  = fmaxf(s0, s1);
  const float e0 = __expf(s0 - m), e1 = __expf(s1 - m);
  const float r  = 1.f / (e0 + e1);
  out[t] = fmaxf(e0 * r * h0[t] + e1 * r * h1[t], 0.f);
}

extern "C" void kernel_launch(void* const* d_in, const int* in_sizes, int n_in,
                              void* d_out, int out_size, void* d_ws, size_t ws_size,
                              hipStream_t stream) {
  // Inputs in setup_inputs() dict order
  const float* x_A = (const float*)d_in[0];
  const float* x_B = (const float*)d_in[1];
  const int*   src[4]; const int* dst[4]; const float* val[4]; const float* a2[4];
  for (int r = 0; r < 4; ++r) {          // 0=AA, 1=AB, 2=BB, 3=BA
    src[r] = (const int*)d_in[2 + 4 * r];
    dst[r] = (const int*)d_in[3 + 4 * r];
    val[r] = (const float*)d_in[4 + 4 * r];
    a2[r]  = (const float*)d_in[5 + 4 * r];
  }
  const float* fc1A_w = (const float*)d_in[18]; const float* fc1A_b = (const float*)d_in[19];
  const float* fc1B_w = (const float*)d_in[20]; const float* fc1B_b = (const float*)d_in[21];
  const float* fcs_w[2] = {(const float*)d_in[22], (const float*)d_in[30]};
  const float* fcs_b[2] = {(const float*)d_in[23], (const float*)d_in[31]};
  const float* saA_w[2] = {(const float*)d_in[24], (const float*)d_in[32]};
  const float* saA_b[2] = {(const float*)d_in[25], (const float*)d_in[33]};
  const float* saA_q[2] = {(const float*)d_in[26], (const float*)d_in[34]};
  const float* saB_w[2] = {(const float*)d_in[27], (const float*)d_in[35]};
  const float* saB_b[2] = {(const float*)d_in[28], (const float*)d_in[36]};
  const float* saB_q[2] = {(const float*)d_in[29], (const float*)d_in[37]};
  const float* fc2_w = (const float*)d_in[38]; const float* fc2_b = (const float*)d_in[39];

  const int N = in_sizes[1] / 128;       // 50000 (x_B is N x 128)
  const int E = in_sizes[2];             // 500000 edges per relation

  const size_t nd = (size_t)N * 128;
  float* A0 = (float*)d_ws;              // xA (current)
  float* A1 = A0 + nd;                   // fcs(xA)
  float* B0 = A1 + nd;                   // xB (current)
  float* B1 = B0 + nd;                   // fcs(xB)
  float* H0 = B1 + nd;                   // relation aggregation 0
  float* H1 = H0 + nd;                   // relation aggregation 1
  float* SS = H1 + nd;                   // 2 attention-score scalars

  const dim3 blk(256);
  const int gemmBlocks = (N + 127) / 128;
  const int ndBlocks   = (int)((nd + 255) / 256);
  const int spmmBlocks = (int)(((long long)E * 32 + 255) / 256);
  const float invN = 1.0f / (float)N;

  // fc1 + relu
  gemm_wmma<8, 1><<<gemmBlocks, blk, 0, stream>>>(x_A, fc1A_w, fc1A_b, A0, nullptr, nullptr, N, 256);
  gemm_wmma<8, 1><<<gemmBlocks, blk, 0, stream>>>(x_B, fc1B_w, fc1B_b, B0, nullptr, nullptr, N, 128);

  for (int hop = 0; hop < 2; ++hop) {
    // shared per-hop linear
    gemm_wmma<8, 0><<<gemmBlocks, blk, 0, stream>>>(A0, fcs_w[hop], fcs_b[hop], A1, nullptr, nullptr, N, 128);
    gemm_wmma<8, 0><<<gemmBlocks, blk, 0, stream>>>(B0, fcs_w[hop], fcs_b[hop], B1, nullptr, nullptr, N, 128);

    // node type A: hAA = spmm(AA, xA) + a2_AA*xA ; hAB = spmm(AB, xB) + a2_AB*xA
    scale_rows<<<ndBlocks, blk, 0, stream>>>(a2[0], A1, H0, (int)nd);
    spmm_edges<<<spmmBlocks, blk, 0, stream>>>(src[0], dst[0], val[0], A1, H0, E);
    scale_rows<<<ndBlocks, blk, 0, stream>>>(a2[1], A1, H1, (int)nd);
    spmm_edges<<<spmmBlocks, blk, 0, stream>>>(src[1], dst[1], val[1], B1, H1, E);
    hipMemsetAsync(SS, 0, 2 * sizeof(float), stream);
    gemm_wmma<8, 2><<<gemmBlocks, blk, 0, stream>>>(H0, saA_w[hop], saA_b[hop], nullptr, saA_q[hop], SS + 0, N, 128);
    gemm_wmma<8, 2><<<gemmBlocks, blk, 0, stream>>>(H1, saA_w[hop], saA_b[hop], nullptr, saA_q[hop], SS + 1, N, 128);
    softmax2_combine<<<ndBlocks, blk, 0, stream>>>(SS, H0, H1, A0, (int)nd, invN);  // xA updated

    // node type B (uses updated xA): hBB = spmm(BB,xB)+a2_BB*xB ; hBA = spmm(BA,xA)+a2_BA*xB
    scale_rows<<<ndBlocks, blk, 0, stream>>>(a2[2], B1, H0, (int)nd);
    spmm_edges<<<spmmBlocks, blk, 0, stream>>>(src[2], dst[2], val[2], B1, H0, E);
    scale_rows<<<ndBlocks, blk, 0, stream>>>(a2[3], B1, H1, (int)nd);
    spmm_edges<<<spmmBlocks, blk, 0, stream>>>(src[3], dst[3], val[3], A0, H1, E);
    hipMemsetAsync(SS, 0, 2 * sizeof(float), stream);
    gemm_wmma<8, 2><<<gemmBlocks, blk, 0, stream>>>(H0, saB_w[hop], saB_b[hop], nullptr, saB_q[hop], SS + 0, N, 128);
    gemm_wmma<8, 2><<<gemmBlocks, blk, 0, stream>>>(H1, saB_w[hop], saB_b[hop], nullptr, saB_q[hop], SS + 1, N, 128);
    softmax2_combine<<<ndBlocks, blk, 0, stream>>>(SS, H0, H1, B0, (int)nd, invN);  // xB updated
  }

  // final projection: out = xA @ fc2_w + fc2_b  (N x 64, no relu)
  gemm_wmma<4, 0><<<gemmBlocks, blk, 0, stream>>>(A0, fc2_w, fc2_b, (float*)d_out, nullptr, nullptr, N, 128);
}
